// MultiheadAttention_35407710388448
// MI455X (gfx1250) — matrix operand order repaired and checked by reading, not compiled, not run on hardware
//
#include <hip/hip_runtime.h>
#include <hip/hip_bf16.h>

typedef __bf16 bf16_t;
typedef __attribute__((ext_vector_type(16))) __bf16 v16bf;
typedef __attribute__((ext_vector_type(8)))  __bf16 v8bf;
typedef __attribute__((ext_vector_type(8)))  float  v8f;
typedef int v4i_vs __attribute__((vector_size(16)));   // int4 for async-LDS builtin

#define SEQLEN 2048
#define BBATCH 2
#define NHEAD  16
#define DHEAD  64
#define DMODEL 1024
#define MROWS  (SEQLEN * BBATCH)   // 4096

// ---------------------------------------------------------------------------
// Fragment loaders (wave32, bf16, 16x16x32 WMMA)
// A-matrix 16x32 (M x K), row-major source with leading dim ld:
//   lanes 0-15 : row M = lane,    K = {0..7, 16..23}
//   lanes16-31 : row M = lane-16, K = {8..15, 24..31}
// ---------------------------------------------------------------------------
__device__ __forceinline__ v16bf load_a_frag(const bf16_t* base, int ld) {
    const int lane = threadIdx.x & 31;
    const bf16_t* p = base + (size_t)(lane & 15) * ld + ((lane >> 4) ? 8 : 0);
    v8bf lo = *(const v8bf*)(p);        // K offsets +0..7 (or +8..15)
    v8bf hi = *(const v8bf*)(p + 16);   // K offsets +16..23 (or +24..31)
    v16bf f;
#pragma unroll
    for (int i = 0; i < 8; ++i) { f[i] = lo[i]; f[8 + i] = hi[i]; }
    return f;
}

// B-matrix 32x16 (K x N), source row-major [N][K] with leading dim ld:
//   lanes 0-15 : col N = lane,     K = {0..7, 16..23}
//   lanes16-31 : col N = lane-16,  K = {8..15, 24..31}
__device__ __forceinline__ v16bf load_b_frag(const bf16_t* base, int ld) {
    const int lane = threadIdx.x & 31;
    const bf16_t* p = base + (size_t)(lane & 15) * ld + ((lane >> 4) ? 8 : 0);
    v8bf lo = *(const v8bf*)(p);
    v8bf hi = *(const v8bf*)(p + 16);
    v16bf f;
#pragma unroll
    for (int i = 0; i < 8; ++i) { f[i] = lo[i]; f[8 + i] = hi[i]; }
    return f;
}

__device__ __forceinline__ v8f wmma_bf16(v16bf a, v16bf b, v8f c) {
    return __builtin_amdgcn_wmma_f32_16x16x32_bf16(
        /*neg_a=*/false, a, /*neg_b=*/false, b,
        /*c_mod=*/(short)0, c, /*reuse_a=*/false, /*reuse_b=*/false);
}

__device__ __forceinline__ float half_max(float v) {
    v = fmaxf(v, __shfl_xor(v, 1, 32));
    v = fmaxf(v, __shfl_xor(v, 2, 32));
    v = fmaxf(v, __shfl_xor(v, 4, 32));
    v = fmaxf(v, __shfl_xor(v, 8, 32));
    return v;
}
__device__ __forceinline__ float half_sum(float v) {
    v += __shfl_xor(v, 1, 32);
    v += __shfl_xor(v, 2, 32);
    v += __shfl_xor(v, 4, 32);
    v += __shfl_xor(v, 8, 32);
    return v;
}

// ---------------------------------------------------------------------------
// Async global->LDS staging (gfx1250 ASYNCcnt path) with sync fallback.
// Builtin signature (from hipcc diagnostic): param0 = int4* addrspace(1).
// ---------------------------------------------------------------------------
#if __has_builtin(__builtin_amdgcn_global_load_async_to_lds_b128)
#define HAS_ASYNC_LDS 1
#else
#define HAS_ASYNC_LDS 0
#endif

__device__ __forceinline__ void stage_b128(const bf16_t* gsrc, bf16_t* ldst) {
#if HAS_ASYNC_LDS
    __builtin_amdgcn_global_load_async_to_lds_b128(
        (__attribute__((address_space(1))) v4i_vs*)gsrc,
        (__attribute__((address_space(3))) v4i_vs*)ldst, 0, 0);
#else
    *(v8bf*)ldst = *(const v8bf*)gsrc;
#endif
}

__device__ __forceinline__ void stage_wait() {
#if HAS_ASYNC_LDS
#if __has_builtin(__builtin_amdgcn_s_wait_asynccnt)
    __builtin_amdgcn_s_wait_asynccnt(0);
#else
    asm volatile("s_wait_asynccnt 0x0" ::: "memory");
#endif
#endif
}

// ---------------------------------------------------------------------------
// fp32 -> bf16 cast
// ---------------------------------------------------------------------------
__global__ void cast_f32_bf16(const float* __restrict__ in,
                              bf16_t* __restrict__ out, int n) {
    int i = blockIdx.x * blockDim.x + threadIdx.x;
    if (i < n) out[i] = (bf16_t)in[i];
}

// ---------------------------------------------------------------------------
// V [s,b,h,d] -> Vt [b,h,d,s]  (so P*V B-operand is contiguous per lane)
// ---------------------------------------------------------------------------
__global__ void transpose_v(const bf16_t* __restrict__ V,
                            bf16_t* __restrict__ Vt) {
    int i = blockIdx.x * blockDim.x + threadIdx.x;
    if (i >= SEQLEN * BBATCH * DMODEL) return;
    int d = i & 63;
    int h = (i >> 6) & 15;
    int b = (i >> 10) & 1;
    int s = i >> 11;
    Vt[(((size_t)(b * NHEAD + h) * DHEAD + d) * SEQLEN) + s] = V[i];
}

// ---------------------------------------------------------------------------
// GEMM: O[M,N] = A[M,K] * W^T (+bias); W row-major [N][K].
// Block = 8 waves x 16 rows (M tile 128), N tile 64.  The 64x32 B tile is
// staged into LDS once per block (double-buffered, async global->LDS) and
// shared by all 8 waves -> 8x less VMEM traffic on the weight stream.
// grid.z selects among up to 3 weight/output sets.
// ---------------------------------------------------------------------------
__global__ void __launch_bounds__(256)
gemm_bf16(const bf16_t* __restrict__ A,
          const bf16_t* __restrict__ W0, const bf16_t* __restrict__ W1,
          const bf16_t* __restrict__ W2,
          const float* __restrict__ bias0, const float* __restrict__ bias1,
          const float* __restrict__ bias2,
          bf16_t* __restrict__ Oh0, bf16_t* __restrict__ Oh1,
          bf16_t* __restrict__ Oh2, float* __restrict__ Of,
          int M, int N, int K, int storeF32) {
    __shared__ bf16_t bsh[2][64 * 32];   // two 4 KB B tiles [n][k]

    const bf16_t* W = W0;
    const float*  bias = bias0;
    bf16_t*       Oh = Oh0;
    if (blockIdx.z == 1) { W = W1; bias = bias1; Oh = Oh1; }
    else if (blockIdx.z == 2) { W = W2; bias = bias2; Oh = Oh2; }

    const int lane = threadIdx.x & 31;
    const int wave = threadIdx.x >> 5;
    const int m0 = blockIdx.y * 128 + wave * 16;
    const int n0 = blockIdx.x * 64;

    // cooperative copy: 256 threads x 16B = one 64x32 bf16 tile per step
    const int tid = threadIdx.x;
    const int cp_n = tid >> 2;            // 0..63
    const int cp_k = (tid & 3) * 8;       // 0,8,16,24
    const bf16_t* gsrc = W + (size_t)(n0 + cp_n) * K + cp_k;
    bf16_t* ldst0 = &bsh[0][cp_n * 32 + cp_k];
    bf16_t* ldst1 = &bsh[1][cp_n * 32 + cp_k];

    v8f zero = {};
    v8f acc[4];
#pragma unroll
    for (int nt = 0; nt < 4; ++nt) acc[nt] = zero;

    // prologue: stage first B tile
    stage_b128(gsrc, ldst0);
    stage_wait();
    __syncthreads();

    const bf16_t* arow = A + (size_t)m0 * K;
    for (int k0 = 0; k0 < K; k0 += 32) {
        const int cur = (k0 >> 5) & 1;
        const bool more = (k0 + 32) < K;
        if (more) {
            stage_b128(gsrc + k0 + 32, cur ? ldst0 : ldst1);
            __builtin_prefetch(arow + k0 + 32, 0, 3);   // near-cache prefetch
        }
        v16bf a = load_a_frag(arow + k0, K);
#pragma unroll
        for (int nt = 0; nt < 4; ++nt) {
            v16bf b = load_b_frag(&bsh[cur][nt * 16 * 32], 32);
            acc[nt] = wmma_bf16(a, b, acc[nt]);
        }
        if (more) stage_wait();
        __syncthreads();
    }

    // C/D layout: vgpr r -> row r (lanes 0-15) / row r+8 (lanes 16-31)
    const int rbase = (lane >> 4) ? 8 : 0;
    const int ncol = lane & 15;
#pragma unroll
    for (int nt = 0; nt < 4; ++nt) {
#pragma unroll
        for (int r = 0; r < 8; ++r) {
            int row = m0 + rbase + r;
            int col = n0 + nt * 16 + ncol;
            float v = acc[nt][r] + bias[col];
            if (storeF32) Of[(size_t)row * N + col] = v;
            else          Oh[(size_t)row * N + col] = (bf16_t)v;
        }
    }
}

// ---------------------------------------------------------------------------
// Flash attention (causal), one wave = 16 query rows for one (b,h).
// Q,K: [s,b,h,d] bf16 (row stride 2048);  Vt: [b,h,d,s] bf16.
// ---------------------------------------------------------------------------
__global__ void __launch_bounds__(128)
flash_attn(const bf16_t* __restrict__ Q, const bf16_t* __restrict__ Kmat,
           const bf16_t* __restrict__ Vt, bf16_t* __restrict__ Ctx) {
    __shared__ bf16_t lds_p[4][16 * 32];   // wave-private 16x32 P tiles

    const int lane = threadIdx.x & 31;
    const int wave = threadIdx.x >> 5;
    const int q0 = blockIdx.x * 64 + wave * 16;
    const int bh = blockIdx.y;
    const int b = bh >> 4, h = bh & 15;

    const int ldq = BBATCH * DMODEL;       // 2048
    const bf16_t* qbase = Q    + (size_t)b * DMODEL + h * DHEAD;
    const bf16_t* kbase = Kmat + (size_t)b * DMODEL + h * DHEAD;
    const bf16_t* vtb   = Vt   + (size_t)bh * DHEAD * SEQLEN;   // [d][s]
    bf16_t*       obase = Ctx  + (size_t)b * DMODEL + h * DHEAD;

    // Q fragments, d in [0,32) and [32,64)
    v16bf qf0 = load_a_frag(qbase + (size_t)q0 * ldq + 0, ldq);
    v16bf qf1 = load_a_frag(qbase + (size_t)q0 * ldq + 32, ldq);

    v8f zero = {};
    v8f acc[4];
#pragma unroll
    for (int nt = 0; nt < 4; ++nt) acc[nt] = zero;
    float mi[8], li[8];
#pragma unroll
    for (int r = 0; r < 8; ++r) { mi[r] = -1e30f; li[r] = 0.0f; }

    const float scale = 0.125f;            // 1/sqrt(64)
    const int rbase = (lane >> 4) ? 8 : 0;
    const int ncol = lane & 15;
    bf16_t* myp = lds_p[wave];

    const int nchunks = (q0 + 15) / 32 + 1;   // causal key-chunk limit
    for (int c = 0; c < nchunks; ++c) {
        const int kb0 = c * 32;

        // --- scores for keys [kb0, kb0+16) and [kb0+16, kb0+32)
        v8f s0 = zero, s1 = zero;
        {
            v16bf kf0 = load_b_frag(kbase + (size_t)kb0 * ldq + 0, ldq);
            v16bf kf1 = load_b_frag(kbase + (size_t)kb0 * ldq + 32, ldq);
            s0 = wmma_bf16(qf0, kf0, s0);
            s0 = wmma_bf16(qf1, kf1, s0);
        }
        {
            v16bf kf0 = load_b_frag(kbase + (size_t)(kb0 + 16) * ldq + 0, ldq);
            v16bf kf1 = load_b_frag(kbase + (size_t)(kb0 + 16) * ldq + 32, ldq);
            s1 = wmma_bf16(qf0, kf0, s1);
            s1 = wmma_bf16(qf1, kf1, s1);
        }

        // --- scale, causal mask, online softmax update
        float mnew[8];
#pragma unroll
        for (int r = 0; r < 8; ++r) {
            const int qi = q0 + rbase + r;
            float a0 = s0[r] * scale;
            float a1 = s1[r] * scale;
            if (kb0 + ncol > qi)      a0 = -1e30f;
            if (kb0 + 16 + ncol > qi) a1 = -1e30f;
            s0[r] = a0; s1[r] = a1;
            mnew[r] = fmaxf(mi[r], half_max(fmaxf(a0, a1)));
        }
#pragma unroll
        for (int r = 0; r < 8; ++r) {
            float p0 = __expf(s0[r] - mnew[r]);
            float p1 = __expf(s1[r] - mnew[r]);
            float rs = half_sum(p0 + p1);
            float corr = __expf(mi[r] - mnew[r]);
            li[r] = li[r] * corr + rs;
            mi[r] = mnew[r];
#pragma unroll
            for (int nt = 0; nt < 4; ++nt) acc[nt][r] = acc[nt][r] * corr;
            const int m = rbase + r;               // C-layout -> LDS row-major
            myp[m * 32 + ncol]      = (bf16_t)p0;
            myp[m * 32 + 16 + ncol] = (bf16_t)p1;
        }

        // --- context: acc += P[16x32] * V[32x64]
        v16bf pf = load_a_frag(myp, 32);
#pragma unroll
        for (int nt = 0; nt < 4; ++nt) {
            v16bf vf = load_b_frag(vtb + (size_t)(nt * 16) * SEQLEN + kb0,
                                   SEQLEN);
            acc[nt] = wmma_bf16(pf, vf, acc[nt]);
        }
    }

    // --- epilogue: normalize, store bf16 context
#pragma unroll
    for (int r = 0; r < 8; ++r) {
        const float inv = 1.0f / li[r];
        const int s = q0 + rbase + r;
#pragma unroll
        for (int nt = 0; nt < 4; ++nt) {
            obase[(size_t)s * ldq + nt * 16 + ncol] =
                (bf16_t)(acc[nt][r] * inv);
        }
    }
}

// ---------------------------------------------------------------------------
extern "C" void kernel_launch(void* const* d_in, const int* in_sizes, int n_in,
                              void* d_out, int out_size, void* d_ws,
                              size_t ws_size, hipStream_t stream) {
    (void)in_sizes; (void)n_in; (void)out_size; (void)ws_size;
    const float* query = (const float*)d_in[0];
    const float* q_w   = (const float*)d_in[1];
    const float* q_b   = (const float*)d_in[2];
    const float* k_w   = (const float*)d_in[3];
    const float* k_b   = (const float*)d_in[4];
    const float* v_w   = (const float*)d_in[5];
    const float* v_b   = (const float*)d_in[6];
    const float* out_w = (const float*)d_in[7];
    const float* out_b = (const float*)d_in[8];
    // d_in[9] = attn_mask: causal, reconstructed analytically.

    char* ws = (char*)d_ws;
    const size_t MB = 1ull << 20;
    bf16_t* qbf = (bf16_t*)(ws);             // 8 MB activations bf16
    bf16_t* wq  = (bf16_t*)(ws + 8 * MB);    // 2 MB each
    bf16_t* wk  = (bf16_t*)(ws + 10 * MB);
    bf16_t* wv  = (bf16_t*)(ws + 12 * MB);
    bf16_t* wo  = (bf16_t*)(ws + 14 * MB);
    bf16_t* Qb  = (bf16_t*)(ws + 16 * MB);   // 8 MB each
    bf16_t* Kb  = (bf16_t*)(ws + 24 * MB);
    bf16_t* Vb  = (bf16_t*)(ws + 32 * MB);
    bf16_t* Vt  = (bf16_t*)(ws + 40 * MB);
    bf16_t* Ctx = (bf16_t*)(ws + 48 * MB);

    const int nAct = MROWS * DMODEL;         // 4,194,304
    const int nW   = DMODEL * DMODEL;        // 1,048,576
    cast_f32_bf16<<<(nAct + 255) / 256, 256, 0, stream>>>(query, qbf, nAct);
    cast_f32_bf16<<<(nW + 255) / 256, 256, 0, stream>>>(q_w, wq, nW);
    cast_f32_bf16<<<(nW + 255) / 256, 256, 0, stream>>>(k_w, wk, nW);
    cast_f32_bf16<<<(nW + 255) / 256, 256, 0, stream>>>(v_w, wv, nW);
    cast_f32_bf16<<<(nW + 255) / 256, 256, 0, stream>>>(out_w, wo, nW);

    // QKV projections (grid.z = which matrix)
    dim3 gqkv(DMODEL / 64, MROWS / 128, 3);
    gemm_bf16<<<gqkv, 256, 0, stream>>>(qbf, wq, wk, wv, q_b, k_b, v_b,
                                        Qb, Kb, Vb, nullptr,
                                        MROWS, DMODEL, DMODEL, 0);

    transpose_v<<<(nAct + 255) / 256, 256, 0, stream>>>(Vb, Vt);

    dim3 gatt(SEQLEN / 64, BBATCH * NHEAD);
    flash_attn<<<gatt, 128, 0, stream>>>(Qb, Kb, Vt, Ctx);

    // output projection -> fp32 d_out
    dim3 gout(DMODEL / 64, MROWS / 128, 1);
    gemm_bf16<<<gout, 256, 0, stream>>>(Ctx, wo, wo, wo, out_b, out_b, out_b,
                                        nullptr, nullptr, nullptr,
                                        (float*)d_out,
                                        MROWS, DMODEL, DMODEL, 1);
}
